// ConvNet3_62603443306831
// MI455X (gfx1250) — matrix operand order repaired and checked by reading, not compiled, not run on hardware
//
#include <hip/hip_runtime.h>
#include <math.h>
#include <stdint.h>

typedef __attribute__((ext_vector_type(16))) _Float16 v16h;
typedef __attribute__((ext_vector_type(8)))  _Float16 v8h;
typedef __attribute__((ext_vector_type(4)))  _Float16 h4;
typedef __attribute__((ext_vector_type(8)))  float    v8f;
typedef float f32x4 __attribute__((ext_vector_type(4), aligned(4)));  // 4B-aligned global loads OK on AMDGPU

#define BATCH   16
#define CCH     64
#define WPTS    125000
#define KH      3
#define KWK     2500
#define KWP     2528      // 79 * 32, zero padded
#define SW      1250
#define TOUT    99
#define NCHUNK  79
#define XROWS   66        // padded channel rows staged per chunk
#define XSTR    40        // LDS row stride in halves (80B, bank-spread)
#define WSTR    40
#define XBUFH   (XROWS * XSTR)
#define WBUFH   (96 * WSTR)

// ---------------- Phase 0: weights fp32 [oc][1][kh][2500] -> f16 [kh*32+oc][2528] (zero pad)
__global__ void prep_weights_k(const float* __restrict__ conv_w, _Float16* __restrict__ wt) {
    int idx = blockIdx.x * 256 + threadIdx.x;
    if (idx >= 96 * KWP) return;
    int row = idx / KWP;            // kh*32 + oc
    int kw  = idx - row * KWP;
    int kh = row >> 5;
    int oc = row & 31;
    float v = (kw < KWK) ? conv_w[(oc * KH + kh) * KWK + kw] : 0.0f;
    wt[idx] = (_Float16)v;
}

// ---------------- Phase 1: fused conv1(+bias,relu) + 1x1 conv(+bias,relu) -> z[t][b][h]
// One WG per (t, b). 4 waves; wave w owns output rows h in [16w, 16w+16), N = 32 oc.
__launch_bounds__(128)
__global__ void conv_fused_k(const float* __restrict__ x,
                             const _Float16* __restrict__ wt,
                             const float* __restrict__ conv_b,
                             const float* __restrict__ conv2_w,
                             const float* __restrict__ conv2_b,
                             float* __restrict__ z)
{
    __shared__ alignas(16) _Float16 xlds[2 * XBUFH];   // double buffered, 10.5 KB
    __shared__ alignas(16) _Float16 wlds[2 * WBUFH];   // double buffered, 15.4 KB

    const int t    = blockIdx.x;
    const int b    = blockIdx.y;
    const int tid  = threadIdx.x;
    const int lane = tid & 31;
    const int wid  = tid >> 5;
    const int hbase = wid << 4;

    const float* xb = x + (size_t)b * CCH * WPTS + (size_t)t * SW;

    // ---- hoisted staging addresses ----
    const int kwg4 = (tid & 7) * 4;     // kw offset within 32-wide chunk (floats), const per thread
    const int r0   = tid >> 3;          // 0..15
    const float* xrow[4];
    int xsoff[4];
#pragma unroll
    for (int j = 0; j < 4; ++j) {
        int row = r0 + 16 * j;          // padded rows 0..63
        int ch  = row - 1;              // replicate pad (low side only needed here)
        ch = ch < 0 ? 0 : ch;
        xrow[j]  = xb + (size_t)ch * WPTS;
        xsoff[j] = row * XSTR + kwg4;
    }
    // tail: padded rows 64,65 both map to channel 63 (row 64 -> ch 63; row 65 -> replicate of 63)
    const float* xrow_t = xb + (size_t)63 * WPTS;
    const int    xsoff_t = (64 + (tid >> 3)) * XSTR + kwg4;   // used only when tid < 16

    const _Float16* wrow[3];
    int wsoff[3];
    const int kwl = (tid & 3) * 8;      // 0,8,16,24 halves
#pragma unroll
    for (int i = 0; i < 3; ++i) {
        int row = (tid >> 2) + 32 * i;  // 0..95 = kh*32+oc
        wrow[i]  = wt + (size_t)row * KWP + kwl;
        wsoff[i] = row * WSTR + kwl;
    }

    v8f c0 = {};
    v8f c1 = {};
    f32x4 xr[4];
    f32x4 xrt;

    // issue fp32 x global loads for chunk c into registers (needs cvt, so via VGPRs)
    auto loadc = [&](int c) {
        int off = c * 32 + kwg4;
        // clamp: K >= 2500 multiplies zero-padded weights, any finite in-bounds value is fine
        off = off > (KWK - 4) ? (KWK - 4) : off;
#pragma unroll
        for (int j = 0; j < 4; ++j) xr[j] = *(const f32x4*)(xrow[j] + off);
        if (tid < 16) xrt = *(const f32x4*)(xrow_t + off);
    };
    // weights are already f16: DMA them global->LDS on the CDNA5 async path (ASYNCcnt),
    // 16B per lane per op; completion consumed by s_wait_asynccnt just before the barrier.
    auto loadw_async = [&](int c, int buf) {
        _Float16* wd = wlds + buf * WBUFH;
#pragma unroll
        for (int i = 0; i < 3; ++i) {
            unsigned lds_addr = (unsigned)(uintptr_t)(wd + wsoff[i]);  // low 32b of flat LDS ptr
            const _Float16* g = wrow[i] + c * 32;
            asm volatile("global_load_async_to_lds_b128 %0, %1, off"
                         :: "v"(lds_addr), "v"(g) : "memory");
        }
    };
    // commit x registers -> LDS buffer `buf`
    auto storec = [&](int buf) {
        _Float16* xd = xlds + buf * XBUFH;
#pragma unroll
        for (int j = 0; j < 4; ++j)
            *(h4*)(xd + xsoff[j]) = __builtin_convertvector(xr[j], h4);
        if (tid < 16)
            *(h4*)(xd + xsoff_t) = __builtin_convertvector(xrt, h4);
    };

    const int ml   = lane & 15;
    const int ksel = (lane < 16) ? 0 : 8;   // A/B: K offset of lane half = 2*ksel halves

    auto compute = [&](int buf) {
        const _Float16* xs = xlds + buf * XBUFH;
        const _Float16* ws = wlds + buf * WBUFH;
#pragma unroll
        for (int kh = 0; kh < KH; ++kh) {
            // A fragment: lane ml = row M; 16 contiguous K halves, lanes>=16 -> K 16..31
            const _Float16* ap = xs + (hbase + kh + ml) * XSTR + 2 * ksel;
            union { v16h v; v8h h[2]; } A;
            A.h[0] = *(const v8h*)(ap);
            A.h[1] = *(const v8h*)(ap + 8);
            // B fragments: lane ml = col N (oc); 16 contiguous K halves, lanes>=16 -> K 16..31
            v16h B0 = *(const v16h*)(ws + (kh * 32 + ml)      * WSTR + 2 * ksel);
            v16h B1 = *(const v16h*)(ws + (kh * 32 + 16 + ml) * WSTR + 2 * ksel);
            c0 = __builtin_amdgcn_wmma_f32_16x16x32_f16(false, A.v, false, B0,
                                                        (short)0, c0, false, false);
            c1 = __builtin_amdgcn_wmma_f32_16x16x32_f16(false, A.v, false, B1,
                                                        (short)0, c1, false, false);
        }
    };

    loadc(0);
    loadw_async(0, 0);
    storec(0);
    asm volatile("s_wait_asynccnt 0x0" ::: "memory");
    __syncthreads();
    for (int c = 0; c < NCHUNK - 1; ++c) {
        loadc(c + 1);                   // fp32 x -> regs, latency covered by WMMAs below
        loadw_async(c + 1, (c + 1) & 1);// f16 weights -> LDS via async DMA, overlapped
        compute(c & 1);                 // consume buffer c&1
        storec((c + 1) & 1);            // commit converted x to the other buffer
        asm volatile("s_wait_asynccnt 0x0" ::: "memory");  // weight DMA done before publish
        __syncthreads();                // single barrier per chunk (double buffered)
    }
    compute((NCHUNK - 1) & 1);

    // Epilogue: bias + relu, 1x1 conv reduce over oc, bias + relu, store z[t][b][h]
    const float cb0 = conv_b[ml];
    const float cb1 = conv_b[16 + ml];
    const float w20 = conv2_w[ml];
    const float w21 = conv2_w[16 + ml];
    const float c2b = conv2_b[0];
#pragma unroll
    for (int r = 0; r < 8; ++r) {
        float p = fmaxf(c0[r] + cb0, 0.0f) * w20 + fmaxf(c1[r] + cb1, 0.0f) * w21;
        // reduce over the 16 oc lanes within each half-wave (xor stays inside 16-block)
        p += __shfl_xor(p, 8, 32);
        p += __shfl_xor(p, 4, 32);
        p += __shfl_xor(p, 2, 32);
        p += __shfl_xor(p, 1, 32);
        if (ml == 0) {
            int h = hbase + r + ((lane >> 4) << 3);   // VGPR r: lanes 0-15 -> M=r, 16-31 -> M=r+8
            z[(size_t)t * (BATCH * CCH) + b * CCH + h] = fmaxf(p + c2b, 0.0f);
        }
    }
}

// ---------------- Phase 2: scalar Elman RNN over T, then sigmoid. 1024 independent lanes.
__global__ void rnn_scan_k(const float* __restrict__ z,
                           const float* __restrict__ w_ih, const float* __restrict__ w_hh,
                           const float* __restrict__ b_ih, const float* __restrict__ b_hh,
                           const float* __restrict__ h0,  float* __restrict__ out)
{
    int idx = blockIdx.x * 256 + threadIdx.x;
    if (idx >= BATCH * CCH) return;
    const float wih  = w_ih[0];
    const float whh  = w_hh[0];
    const float bias = b_ih[0] + b_hh[0];
    float h = h0[idx >> 6];                 // h0[b] broadcast over channels
    for (int tt = 0; tt < TOUT; ++tt)
        h = tanhf(wih * z[(size_t)tt * (BATCH * CCH) + idx] + whh * h + bias);
    out[idx] = 1.0f / (1.0f + expf(-h));
}

extern "C" void kernel_launch(void* const* d_in, const int* in_sizes, int n_in,
                              void* d_out, int out_size, void* d_ws, size_t ws_size,
                              hipStream_t stream) {
    const float* x       = (const float*)d_in[0];
    const float* conv_w  = (const float*)d_in[1];
    const float* conv_b  = (const float*)d_in[2];
    const float* conv2_w = (const float*)d_in[3];
    const float* conv2_b = (const float*)d_in[4];
    const float* w_ih    = (const float*)d_in[5];
    const float* w_hh    = (const float*)d_in[6];
    const float* b_ih    = (const float*)d_in[7];
    const float* b_hh    = (const float*)d_in[8];
    const float* h0      = (const float*)d_in[9];

    _Float16* wt = (_Float16*)d_ws;                                   // 96*2528*2 = 485376 B
    float*    z  = (float*)((char*)d_ws + (size_t)96 * KWP * 2);      // 99*1024*4 = 405504 B

    prep_weights_k<<<(96 * KWP + 255) / 256, 256, 0, stream>>>(conv_w, wt);

    dim3 grid(TOUT, BATCH);
    conv_fused_k<<<grid, 128, 0, stream>>>(x, wt, conv_b, conv2_w, conv2_b, z);

    rnn_scan_k<<<(BATCH * CCH + 255) / 256, 256, 0, stream>>>(z, w_ih, w_hh, b_ih, b_hh, h0,
                                                              (float*)d_out);
}
